// PhiFlashAttention2_22789096472817
// MI455X (gfx1250) — matrix operand order repaired and checked by reading, not compiled, not run on hardware
//
#include <hip/hip_runtime.h>
#include <hip/hip_bf16.h>

// Problem constants (from reference)
#define S    2048
#define Hd   2048     // hidden
#define NH   32
#define HD   64

typedef __attribute__((ext_vector_type(16))) __bf16 v16bf;
typedef __attribute__((ext_vector_type(8)))  __bf16 v8bf;
typedef __attribute__((ext_vector_type(8)))  float  v8f;

__device__ __forceinline__ v16bf cat8(v8bf lo, v8bf hi) {
    return __builtin_shufflevector(lo, hi, 0,1,2,3,4,5,6,7,8,9,10,11,12,13,14,15);
}

__device__ __forceinline__ v8f wmma_bf16(v16bf a, v16bf b, v8f c) {
    // D(f32,16x16) = A(bf16,16x32) * B(bf16,32x16) + C
    return __builtin_amdgcn_wmma_f32_16x16x32_bf16(false, a, false, b, (short)0, c, false, false);
}

// ---------------------------------------------------------------------------
// fp32 -> bf16 conversion
// ---------------------------------------------------------------------------
__global__ void f32_to_bf16_kernel(const float* __restrict__ in, __bf16* __restrict__ out, int n) {
    int i = blockIdx.x * blockDim.x + threadIdx.x;
    if (i < n) out[i] = (__bf16)in[i];
}

// ---------------------------------------------------------------------------
// GEMM: out[M,N] = A[M,K] @ Bt[N,K]^T + bias[N]
// Block: 256 threads = 8 wave32, block tile 128x128, wave tile 32x64.
// A-frag (16x32 bf16): lane L, row = L%16; 16 contiguous k at
//   k+(L>=16?16:0) -> two 16B contiguous loads.
// B-frag (32x16 bf16): lane L, col = L%16; 16 contiguous k at
//   k+(L>=16?16:0) -> one 32B contiguous load.
// ---------------------------------------------------------------------------
__global__ __launch_bounds__(256)
void gemm_bf16_kernel(const __bf16* __restrict__ A, const __bf16* __restrict__ Bt,
                      const float* __restrict__ bias,
                      float* __restrict__ outF, __bf16* __restrict__ outB,
                      int M, int N, int K)
{
    const int wave = threadIdx.x >> 5, lane = threadIdx.x & 31;
    const int ln   = lane & 15,        hi16 = (lane >> 4) & 1;
    const int wm   = wave & 3,         wn   = wave >> 2;
    const int rowBase = blockIdx.y * 128 + wm * 32;
    const int colBase = blockIdx.x * 128 + wn * 64;
    const int aoff = hi16 ? 16 : 0;
    const int boff = hi16 ? 16 : 0;

    v8f acc[2][4];
#pragma unroll
    for (int nt = 0; nt < 4; ++nt) {
        float bv = bias[colBase + nt * 16 + ln];
#pragma unroll
        for (int mt = 0; mt < 2; ++mt)
#pragma unroll
            for (int r = 0; r < 8; ++r) acc[mt][nt][r] = bv;
    }

    const __bf16* aRow0 = A + (size_t)(rowBase + ln) * K;
    const __bf16* aRow1 = A + (size_t)(rowBase + 16 + ln) * K;
    const __bf16* bRow0 = Bt + (size_t)(colBase + 0 * 16 + ln) * K;
    const __bf16* bRow1 = Bt + (size_t)(colBase + 1 * 16 + ln) * K;
    const __bf16* bRow2 = Bt + (size_t)(colBase + 2 * 16 + ln) * K;
    const __bf16* bRow3 = Bt + (size_t)(colBase + 3 * 16 + ln) * K;

    for (int k = 0; k < K; k += 32) {
        v16bf a0 = cat8(*(const v8bf*)(aRow0 + k + aoff), *(const v8bf*)(aRow0 + k + 8 + aoff));
        v16bf a1 = cat8(*(const v8bf*)(aRow1 + k + aoff), *(const v8bf*)(aRow1 + k + 8 + aoff));
        v16bf b0 = *(const v16bf*)(bRow0 + k + boff);
        v16bf b1 = *(const v16bf*)(bRow1 + k + boff);
        v16bf b2 = *(const v16bf*)(bRow2 + k + boff);
        v16bf b3 = *(const v16bf*)(bRow3 + k + boff);
        acc[0][0] = wmma_bf16(a0, b0, acc[0][0]);
        acc[1][0] = wmma_bf16(a1, b0, acc[1][0]);
        acc[0][1] = wmma_bf16(a0, b1, acc[0][1]);
        acc[1][1] = wmma_bf16(a1, b1, acc[1][1]);
        acc[0][2] = wmma_bf16(a0, b2, acc[0][2]);
        acc[1][2] = wmma_bf16(a1, b2, acc[1][2]);
        acc[0][3] = wmma_bf16(a0, b3, acc[0][3]);
        acc[1][3] = wmma_bf16(a1, b3, acc[1][3]);
    }

    // C layout: VGPR r holds row (r + hi16*8), lane%16 = col
#pragma unroll
    for (int mt = 0; mt < 2; ++mt)
#pragma unroll
        for (int nt = 0; nt < 4; ++nt)
#pragma unroll
            for (int r = 0; r < 8; ++r) {
                int row = rowBase + mt * 16 + r + hi16 * 8;
                int col = colBase + nt * 16 + ln;
                float v = acc[mt][nt][r];
                if (outB) outB[(size_t)row * N + col] = (__bf16)v;
                else      outF[(size_t)row * N + col] = v;
            }
}

// ---------------------------------------------------------------------------
// RoPE in-place on bf16 Q and K. One thread per (s, head, d<32) pair.
// ---------------------------------------------------------------------------
__global__ void rope_kernel(__bf16* __restrict__ Q, __bf16* __restrict__ K) {
    int idx = blockIdx.x * blockDim.x + threadIdx.x;   // S*NH*32 threads
    int d = idx & 31;
    int h = (idx >> 5) & (NH - 1);
    int s = idx >> 10;
    float inv = __powf(10000.0f, -(2.0f * (float)d) / (float)HD);
    float ang = (float)s * inv;
    float c = __cosf(ang), sn = __sinf(ang);
    size_t base = (size_t)s * Hd + h * HD;
    float q1 = (float)Q[base + d], q2 = (float)Q[base + d + 32];
    Q[base + d]      = (__bf16)(q1 * c - q2 * sn);
    Q[base + d + 32] = (__bf16)(q2 * c + q1 * sn);
    float k1 = (float)K[base + d], k2 = (float)K[base + d + 32];
    K[base + d]      = (__bf16)(k1 * c - k2 * sn);
    K[base + d + 32] = (__bf16)(k2 * c + k1 * sn);
}

// ---------------------------------------------------------------------------
// Causal flash attention. grid = (NH, S/64). Block = 128 threads (4 waves),
// each wave owns 16 query rows. Online softmax fully in registers via
// __shfl_xor row reductions; P goes through LDS to re-layout C->A.
// ---------------------------------------------------------------------------
#define VP 72   // LDS pitch (halves): 144B rows -> 16B aligned, spreads banks

__global__ __launch_bounds__(128)
void attn_kernel(const __bf16* __restrict__ Qm, const __bf16* __restrict__ Km,
                 const __bf16* __restrict__ Vm, __bf16* __restrict__ Om)
{
    __shared__ __bf16 sVt[64 * VP];   // V^T tile: sVt[d*VP + kpos]
    __shared__ __bf16 sP [64 * VP];   // P tile:   sP[qrow*VP + kpos]

    const int h    = blockIdx.x;
    const int qb   = blockIdx.y * 64;
    const int wave = threadIdx.x >> 5, lane = threadIdx.x & 31;
    const int ln   = lane & 15,        hi16 = (lane >> 4) & 1;
    const int aoff = hi16 ? 16 : 0;
    const int boff = hi16 ? 16 : 0;

    // Persistent Q A-fragments (contraction dim = HD, two 32-chunks)
    const __bf16* qrow = Qm + (size_t)(qb + wave * 16 + ln) * Hd + h * HD;
    v16bf aQ[2];
    aQ[0] = cat8(*(const v8bf*)(qrow + 0  + aoff), *(const v8bf*)(qrow + 8 + aoff));
    aQ[1] = cat8(*(const v8bf*)(qrow + 32 + aoff), *(const v8bf*)(qrow + 40 + aoff));

    float mrow[8], lrow[8];
    v8f Oa[4] = {};
#pragma unroll
    for (int r = 0; r < 8; ++r) { mrow[r] = -1e30f; lrow[r] = 0.0f; }

    for (int kb = 0; kb <= qb; kb += 64) {
        // Stage V^T into LDS (coalesced global read, strided LDS write)
        for (int t = threadIdx.x; t < 64 * 64; t += 128) {
            int d = t & 63, j = t >> 6;
            sVt[d * VP + j] = Vm[(size_t)(kb + j) * Hd + h * HD + d];
        }
        __syncthreads();

        // scores = Q @ K^T : B-frag cols = key positions, contraction = d
        v8f accS[4] = {};
#pragma unroll
        for (int c = 0; c < 2; ++c)
#pragma unroll
            for (int nt = 0; nt < 4; ++nt) {
                const __bf16* kp = Km + (size_t)(kb + nt * 16 + ln) * Hd + h * HD + c * 32 + boff;
                accS[nt] = wmma_bf16(aQ[c], *(const v16bf*)kp, accS[nt]);
            }

        // --- online softmax (registers only) ---
        float sv[4][8];
#pragma unroll
        for (int nt = 0; nt < 4; ++nt)
#pragma unroll
            for (int r = 0; r < 8; ++r) {
                float x = accS[nt][r] * 0.125f;           // 1/sqrt(64)
                if (kb == qb) {                            // diagonal block mask
                    int col = kb + nt * 16 + ln;
                    int row = qb + wave * 16 + r + hi16 * 8;
                    if (col > row) x = -1e30f;
                }
                sv[nt][r] = x;
            }
        float mnew[8], al[8];
#pragma unroll
        for (int r = 0; r < 8; ++r) {
            float t = fmaxf(fmaxf(sv[0][r], sv[1][r]), fmaxf(sv[2][r], sv[3][r]));
            t = fmaxf(t, __shfl_xor(t, 1, 32));
            t = fmaxf(t, __shfl_xor(t, 2, 32));
            t = fmaxf(t, __shfl_xor(t, 4, 32));
            t = fmaxf(t, __shfl_xor(t, 8, 32));
            mnew[r] = fmaxf(mrow[r], t);
            al[r]   = __expf(mrow[r] - mnew[r]);
            mrow[r] = mnew[r];
        }
#pragma unroll
        for (int nt = 0; nt < 4; ++nt)
#pragma unroll
            for (int r = 0; r < 8; ++r) sv[nt][r] = __expf(sv[nt][r] - mnew[r]);
#pragma unroll
        for (int r = 0; r < 8; ++r) {
            float ts = sv[0][r] + sv[1][r] + sv[2][r] + sv[3][r];
            ts += __shfl_xor(ts, 1, 32);
            ts += __shfl_xor(ts, 2, 32);
            ts += __shfl_xor(ts, 4, 32);
            ts += __shfl_xor(ts, 8, 32);
            lrow[r] = lrow[r] * al[r] + ts;
        }
#pragma unroll
        for (int nt = 0; nt < 4; ++nt)
#pragma unroll
            for (int r = 0; r < 8; ++r) Oa[nt][r] *= al[r];

        // P: C-layout -> LDS -> A-layout (wave-local region of sP)
#pragma unroll
        for (int nt = 0; nt < 4; ++nt)
#pragma unroll
            for (int r = 0; r < 8; ++r)
                sP[(wave * 16 + r + hi16 * 8) * VP + nt * 16 + ln] = (__bf16)sv[nt][r];
        // wave-local LDS RAW: LDS completes in-order per wave; fence compiler +
        // wait the CDNA5 DS counter explicitly.
        asm volatile("s_wait_dscnt 0" ::: "memory");

        // O += P @ V : contraction = key position (two 32-chunks)
#pragma unroll
        for (int c2 = 0; c2 < 2; ++c2) {
            const __bf16* pr = &sP[(wave * 16 + ln) * VP + c2 * 32];
            v16bf aP = cat8(*(const v8bf*)(pr + aoff), *(const v8bf*)(pr + 8 + aoff));
#pragma unroll
            for (int nt = 0; nt < 4; ++nt) {
                const __bf16* vr = &sVt[(nt * 16 + ln) * VP + c2 * 32 + boff];
                v16bf bV = cat8(*(const v8bf*)vr, *(const v8bf*)(vr + 8));
                Oa[nt] = wmma_bf16(aP, bV, Oa[nt]);
            }
        }
        __syncthreads();   // before next iteration overwrites sVt
    }

    // epilogue: divide by softmax denom, store bf16 into attention-out scratch
#pragma unroll
    for (int nt = 0; nt < 4; ++nt)
#pragma unroll
        for (int r = 0; r < 8; ++r) {
            int row = qb + wave * 16 + r + hi16 * 8;
            float v = Oa[nt][r] / lrow[r];
            Om[(size_t)row * Hd + h * HD + nt * 16 + ln] = (__bf16)v;
        }
}

// ---------------------------------------------------------------------------
// Launch
// ---------------------------------------------------------------------------
extern "C" void kernel_launch(void* const* d_in, const int* in_sizes, int n_in,
                              void* d_out, int out_size, void* d_ws, size_t ws_size,
                              hipStream_t stream) {
    const float* x  = (const float*)d_in[0];
    const float* wq = (const float*)d_in[1];
    const float* bq = (const float*)d_in[2];
    const float* wk = (const float*)d_in[3];
    const float* bk = (const float*)d_in[4];
    const float* wv = (const float*)d_in[5];
    const float* bv = (const float*)d_in[6];
    const float* wo = (const float*)d_in[7];
    const float* bo = (const float*)d_in[8];
    float* out = (float*)d_out;

    const size_t NE = (size_t)S * Hd;            // 4M elements per matrix
    char* ws = (char*)d_ws;
    __bf16* xb  = (__bf16*)(ws);
    __bf16* wqb = xb  + NE;
    __bf16* wkb = wqb + NE;
    __bf16* wvb = wkb + NE;
    __bf16* wob = wvb + NE;
    __bf16* Qb  = wob + NE;
    __bf16* Kb  = Qb  + NE;
    __bf16* Vb  = Kb  + NE;
    __bf16* Ab  = Vb  + NE;                      // total 9 * 8MB = 72MB

    const int n = (int)NE;
    dim3 cgrid((n + 255) / 256);
    f32_to_bf16_kernel<<<cgrid, 256, 0, stream>>>(x,  xb,  n);
    f32_to_bf16_kernel<<<cgrid, 256, 0, stream>>>(wq, wqb, n);
    f32_to_bf16_kernel<<<cgrid, 256, 0, stream>>>(wk, wkb, n);
    f32_to_bf16_kernel<<<cgrid, 256, 0, stream>>>(wv, wvb, n);
    f32_to_bf16_kernel<<<cgrid, 256, 0, stream>>>(wo, wob, n);

    dim3 ggrid(Hd / 128, S / 128);               // 16 x 16
    gemm_bf16_kernel<<<ggrid, 256, 0, stream>>>(xb, wqb, bq, nullptr, Qb, S, Hd, Hd);
    gemm_bf16_kernel<<<ggrid, 256, 0, stream>>>(xb, wkb, bk, nullptr, Kb, S, Hd, Hd);
    gemm_bf16_kernel<<<ggrid, 256, 0, stream>>>(xb, wvb, bv, nullptr, Vb, S, Hd, Hd);

    rope_kernel<<<(S * NH * 32) / 256, 256, 0, stream>>>(Qb, Kb);

    attn_kernel<<<dim3(NH, S / 64), 128, 0, stream>>>(Qb, Kb, Vb, Ab);

    gemm_bf16_kernel<<<ggrid, 256, 0, stream>>>(Ab, wob, bo, out, nullptr, S, Hd, Hd);
}
